// NEGLoss_8358006358140
// MI455X (gfx1250) — compile-verified, matrix operand order checked by reading, not run on hardware
//
#include <hip/hip_runtime.h>
#include <math.h>

#define NEG_V    50257
#define NEG_B    8192
#define NEG_NNEG 5

typedef __attribute__((ext_vector_type(2)))  float    v2f;
typedef __attribute__((ext_vector_type(8)))  float    v8f;
typedef __attribute__((ext_vector_type(16))) _Float16 v16h;

// ---------------------------------------------------------------- RNG helpers
__device__ __forceinline__ unsigned long long neg_splitmix64(unsigned long long x) {
  x += 0x9E3779B97F4A7C15ull;
  x = (x ^ (x >> 30)) * 0xBF58476D1CE4E5B9ull;
  x = (x ^ (x >> 27)) * 0x94D049BB133111EBull;
  return x ^ (x >> 31);
}

// Order-preserving float -> uint key (total order, -inf smallest)
__device__ __forceinline__ unsigned int neg_float_key(float f) {
  unsigned int b = __float_as_uint(f);
  return (b & 0x80000000u) ? ~b : (b | 0x80000000u);
}

// ------------------------------------------------- WMMA wave-sum (wave32 only)
// Returns: lanes 0..15 hold sum of rows 0..7, lanes 16..31 hold rows 8..15 of
// D = A x ones + 0, where A row m = val(lane m) + val(lane m+16).
// So (lane0 result + lane16 result) == sum over all 32 lanes.
__device__ __forceinline__ float neg_wave_partial_wmma(float val) {
  v8f c = {};
#if __has_builtin(__builtin_amdgcn_wmma_f32_16x16x4_f32)
  v2f a; a.x = val; a.y = 0.0f;
  v2f b; b.x = 1.0f; b.y = 1.0f;
  c = __builtin_amdgcn_wmma_f32_16x16x4_f32(false, a, false, b, (short)0, c,
                                            false, false);
#else
  v16h a = {};
  a.s0 = (_Float16)val;
  const _Float16 one = (_Float16)1.0f;
  v16h b = {one, one, one, one, one, one, one, one,
            one, one, one, one, one, one, one, one};
  c = __builtin_amdgcn_wmma_f32_16x16x32_f16(false, a, false, b, (short)0, c,
                                             false, false);
#endif
  return c.s0 + c.s1 + c.s2 + c.s3 + c.s4 + c.s5 + c.s6 + c.s7;
}

// ------------------------------------------------------------------- kernels
// K1: zero weights[V], 5 argmax slots, 2 accumulators
__global__ void neg_init(float* __restrict__ weights,
                         unsigned long long* __restrict__ slots,
                         float* __restrict__ acc) {
  int v = blockIdx.x * blockDim.x + threadIdx.x;
  if (v < NEG_V)    weights[v] = 0.0f;
  if (v < NEG_NNEG) slots[v]   = 0ull;
  if (v < 2)        acc[v]     = 0.0f;
}

// K2: weights[target[i]] += 1
__global__ void neg_scatter(const int* __restrict__ target,
                            float* __restrict__ weights) {
  int i = blockIdx.x * blockDim.x + threadIdx.x;
  if (i < NEG_B) atomicAdd(&weights[target[i]], 1.0f);
}

// K3: Gumbel-max categorical over masked freq^0.75 (norm factor irrelevant).
//     score_k(v) = 0.75*log(wf[v]) + Gumbel(hash(42,k,v)); positives -> -inf.
__global__ void neg_sample(const float* __restrict__ wf,
                           const float* __restrict__ weights,
                           unsigned long long* __restrict__ slots) {
  int v = blockIdx.x * blockDim.x + threadIdx.x;
  if (v >= NEG_V) return;
  float logp = (weights[v] > 0.0f) ? -INFINITY : 0.75f * logf(wf[v]);
#pragma unroll
  for (int k = 0; k < NEG_NNEG; ++k) {
    unsigned long long h = neg_splitmix64(
        (unsigned long long)v * (unsigned long long)NEG_NNEG +
        (unsigned long long)k + (42ull << 32));
    float u = ((float)(unsigned int)(h >> 40) + 0.5f) * (1.0f / 16777216.0f);
    float g = -logf(-logf(u));                 // Gumbel(0,1); +inf if u rounds to 1.0f
    float s = logp + g;
    unsigned long long packed =
        ((unsigned long long)neg_float_key(s) << 32) | (unsigned int)v;
    atomicMax(&slots[k], packed);              // GLOBAL_ATOMIC_MAX_U64
  }
}

// K4: weights[neg_k] += 1
__global__ void neg_add_negatives(const unsigned long long* __restrict__ slots,
                                  float* __restrict__ weights) {
  int k = threadIdx.x;
  if (k < NEG_NNEG) {
    unsigned int idx = (unsigned int)(slots[k] & 0xFFFFFFFFull);
    atomicAdd(&weights[idx], 1.0f);
  }
}

// K5: num += w[t_i]*input[i,t_i], den += w[t_i]; wave-reduced via V_WMMA,
//     lanes 0 and 16 carry the two half-sums -> 2 atomics per wave per acc.
//     Launched with exactly NEG_B threads: EXEC is all ones at the WMMA.
__global__ void neg_loss_reduce(const float* __restrict__ input,
                                const int* __restrict__ target,
                                const float* __restrict__ weights,
                                float* __restrict__ acc) {
  int i = blockIdx.x * blockDim.x + threadIdx.x;
  int t = target[i];
  float w = weights[t];
  float x = input[(size_t)i * (size_t)NEG_V + (size_t)t];
  float np = neg_wave_partial_wmma(w * x);
  float dp = neg_wave_partial_wmma(w);
  unsigned int lane = threadIdx.x & 31u;
  if (lane == 0u || lane == 16u) {
    atomicAdd(&acc[0], np);
    atomicAdd(&acc[1], dp);
  }
}

// K6: out = -num/den
__global__ void neg_finalize(const float* __restrict__ acc,
                             float* __restrict__ out) {
  out[0] = -acc[0] / acc[1];
}

// ------------------------------------------------------------------ launcher
extern "C" void kernel_launch(void* const* d_in, const int* in_sizes, int n_in,
                              void* d_out, int out_size, void* d_ws,
                              size_t ws_size, hipStream_t stream) {
  const float* input  = (const float*)d_in[0];  // [B, V] f32
  const int*   target = (const int*)d_in[1];    // [B] int
  const float* wf     = (const float*)d_in[2];  // [V] f32
  float* out = (float*)d_out;

  char* ws = (char*)d_ws;
  float* weights = (float*)ws;
  size_t off = (((size_t)NEG_V * sizeof(float)) + 7u) & ~(size_t)7u;
  unsigned long long* slots = (unsigned long long*)(ws + off);
  float* acc = (float*)(ws + off + NEG_NNEG * sizeof(unsigned long long));

  const int tb = 256;
  neg_init<<<(NEG_V + tb - 1) / tb, tb, 0, stream>>>(weights, slots, acc);
  neg_scatter<<<(NEG_B + tb - 1) / tb, tb, 0, stream>>>(target, weights);
  neg_sample<<<(NEG_V + tb - 1) / tb, tb, 0, stream>>>(wf, weights, slots);
  neg_add_negatives<<<1, 32, 0, stream>>>(slots, weights);
  neg_loss_reduce<<<NEG_B / tb, tb, 0, stream>>>(input, target, weights, acc);
  neg_finalize<<<1, 1, 0, stream>>>(acc, out);
}
